// Network_27238682591815
// MI455X (gfx1250) — matrix operand order repaired and checked by reading, not compiled, not run on hardware
//
#include <hip/hip_runtime.h>
#include <stdint.h>

#define B_   64
#define L_   512
#define D_   768
#define J_   31
#define S_   32
#define H_   100
#define NC_  10
#define EMB_ 50
#define G3_  300   // 3*H

typedef __attribute__((ext_vector_type(16))) __bf16 v16bf;
typedef __attribute__((ext_vector_type(8)))  float  v8f;

__device__ __forceinline__ __bf16 f2bf(float f) {
  union { float f; uint32_t u; } v; v.f = f;
  uint32_t r = v.u + 0x7FFFu + ((v.u >> 16) & 1u);   // round-to-nearest-even
  union { uint16_t s; __bf16 b; } o; o.s = (uint16_t)(r >> 16);
  return o.b;
}

// ---------------------------------------------------------------------------
// K0: fp32 -> bf16 weight conversion
// ---------------------------------------------------------------------------
__global__ __launch_bounds__(256) void cvt_bf16_kernel(
    const float* __restrict__ src, __bf16* __restrict__ dst, int n) {
  int i = blockIdx.x * 256 + threadIdx.x;
  if (i < n) dst[i] = f2bf(src[i]);
}

// ---------------------------------------------------------------------------
// K1: per-clause softmax pooling. One block per (b, s). Segments are
// contiguous (boundaries strictly increasing), so seg s = [c[s-1], c[s]).
// Writes doc as bf16 in [t*B + b, D] layout (GEMM-ready).
// ---------------------------------------------------------------------------
__global__ __launch_bounds__(256) void pool_kernel(
    const float* __restrict__ hidden, const int* __restrict__ cb,
    const float* __restrict__ fw, const float* __restrict__ fb,
    __bf16* __restrict__ doc)
{
  __shared__ float sc[L_];
  __shared__ float red[256];
  __shared__ float sM, sSum;
  const int b = blockIdx.x / S_, s = blockIdx.x % S_;
  const int start = (s == 0)  ? 0  : cb[b * J_ + s - 1];
  const int end   = (s == J_) ? L_ : cb[b * J_ + s];
  const int len = end - start;
  const int tid = threadIdx.x, lane = tid & 31, wv = tid >> 5;

  // scores: one wave per row, lane-parallel dot of length 768 + shfl reduce
  for (int lr = wv; lr < len; lr += 8) {
    const float* row = hidden + ((size_t)b * L_ + start + lr) * D_;
    float acc = 0.f;
    for (int k = lane; k < D_; k += 32) acc += row[k] * fw[k];
    for (int o = 16; o > 0; o >>= 1) acc += __shfl_down(acc, o);
    if (lane == 0) sc[lr] = acc + fb[0];
  }
  __syncthreads();

  // block max
  float m = -3.4e38f;
  for (int l = tid; l < len; l += 256) m = fmaxf(m, sc[l]);
  red[tid] = m; __syncthreads();
  for (int o = 128; o > 0; o >>= 1) {
    if (tid < o) red[tid] = fmaxf(red[tid], red[tid + o]);
    __syncthreads();
  }
  if (tid == 0) sM = red[0];
  __syncthreads();

  // block sum of exp
  float sum = 0.f;
  for (int l = tid; l < len; l += 256) sum += expf(sc[l] - sM);
  red[tid] = sum; __syncthreads();
  for (int o = 128; o > 0; o >>= 1) {
    if (tid < o) red[tid] += red[tid + o];
    __syncthreads();
  }
  if (tid == 0) sSum = red[0];
  __syncthreads();

  const float inv = 1.f / sSum;
  for (int l = tid; l < len; l += 256) sc[l] = expf(sc[l] - sM) * inv;
  __syncthreads();

  // weighted sum over the segment, emit bf16
  const float* base = hidden + ((size_t)b * L_ + start) * D_;
  for (int d = tid; d < D_; d += 256) {
    float acc = 0.f;
    for (int l = 0; l < len; ++l) acc += sc[l] * base[(size_t)l * D_ + d];
    doc[((size_t)s * B_ + b) * D_ + d] = f2bf(acc);
  }
}

// ---------------------------------------------------------------------------
// K2: gi = doc @ W_ih^T + b_ih via v_wmma_f32_16x16x32_bf16.
// One wave per 16x16 output tile; K loop = 768/32 = 24 WMMA ops.
// A layout (16x32 bf16, ISA 7.12.2): half-wave hi gets K packs
//   [k0+8*hi .. +7] and [k0+16+8*hi .. +7] for row M = lane&15.
// B layout (32x16 bf16): half-wave hi gets K packs
//   [k0+8*hi .. +7] and [k0+16+8*hi .. +7] for column N = lane&15.
// C layout: vgpr r -> M = r + 8*hi, N = lane&15.
// ---------------------------------------------------------------------------
__global__ __launch_bounds__(128) void gi_wmma_kernel(
    const __bf16* __restrict__ A,    // [S*B = 2048, 768]
    const __bf16* __restrict__ W,    // [300, 768]
    const float*  __restrict__ bias, // [300]
    float* __restrict__ C)           // [2048, 300]
{
  const int lane  = threadIdx.x & 31;
  const int wv    = threadIdx.x >> 5;
  const int mtile = blockIdx.x * 4 + wv;   // 0..127
  const int ntile = blockIdx.y;            // 0..18 (300 cols, last tile partial)
  const int hi    = lane >> 4;
  const int r16   = lane & 15;
  const int arow  = mtile * 16 + r16;
  const int wrow  = ntile * 16 + r16;
  const int wrow_ld = (wrow < G3_) ? wrow : 0;   // clamp OOB loads; never stored

  const __bf16* ap = A + (size_t)arow * D_ + hi * 8;
  const __bf16* bp = W + (size_t)wrow_ld * D_ + hi * 8;

  v8f c = {};
#pragma unroll 4
  for (int k0 = 0; k0 < D_; k0 += 32) {
    union { v16bf v; uint4 q[2]; } a, bm;
    a.q[0]  = *reinterpret_cast<const uint4*>(ap + k0);
    a.q[1]  = *reinterpret_cast<const uint4*>(ap + k0 + 16);
    bm.q[0] = *reinterpret_cast<const uint4*>(bp + k0);
    bm.q[1] = *reinterpret_cast<const uint4*>(bp + k0 + 16);
    c = __builtin_amdgcn_wmma_f32_16x16x32_bf16(
            /*neg_a=*/false, a.v, /*neg_b=*/false, bm.v,
            /*c_mod=*/(short)0, c, /*reuse_a=*/false, /*reuse_b=*/false);
  }

  const int col = ntile * 16 + r16;
  if (col < G3_) {
    const float bs = bias[col];
#pragma unroll
    for (int r = 0; r < 8; ++r) {
      const int m = mtile * 16 + r + hi * 8;
      C[(size_t)m * G3_ + col] = c[r] + bs;
    }
  }
}

// ---------------------------------------------------------------------------
// K3: encoder GRU recurrence, both directions concurrently (grid.y = dir).
// W_hh (300x100 fp32 = 120 KB) cached in dynamic LDS; 32 sequential steps.
// gi already includes b_ih.
// ---------------------------------------------------------------------------
__global__ __launch_bounds__(320) void enc_gru_kernel(
    const float* __restrict__ gi_f, const float* __restrict__ gi_b,
    const float* __restrict__ whh_f, const float* __restrict__ whh_b,
    const float* __restrict__ bhh_f, const float* __restrict__ bhh_b,
    float* __restrict__ ys_f, float* __restrict__ ys_b,
    float* __restrict__ hT_f)
{
  extern __shared__ float sm[];
  float* sW  = sm;            // 300*100
  float* sH  = sm + 30000;    // 100
  float* sGh = sm + 30112;    // 300
  float* sHn = sm + 30416;    // 100   (total 30516 floats)
  const int b = blockIdx.x, dir = blockIdx.y, tid = threadIdx.x;
  const float* gi  = dir ? gi_b  : gi_f;
  const float* whh = dir ? whh_b : whh_f;
  const float* bhh = dir ? bhh_b : bhh_f;
  float*       ys  = dir ? ys_b  : ys_f;

  for (int i = tid; i < G3_ * H_; i += 320) sW[i] = whh[i];
  if (tid < H_) sH[tid] = 0.f;
  __syncthreads();

  for (int step = 0; step < S_; ++step) {
    const int t = dir ? (S_ - 1 - step) : step;
    const float* g = gi + ((size_t)t * B_ + b) * G3_;
    if (tid < G3_) {
      const float* wr = sW + tid * H_;
      float acc = bhh[tid];
      for (int k = 0; k < H_; ++k) acc += wr[k] * sH[k];
      sGh[tid] = acc;
    }
    __syncthreads();
    if (tid < H_) {
      const float r = 1.f / (1.f + expf(-(g[tid]        + sGh[tid])));
      const float z = 1.f / (1.f + expf(-(g[H_ + tid]   + sGh[H_ + tid])));
      const float n = tanhf(g[2 * H_ + tid] + r * sGh[2 * H_ + tid]);
      sHn[tid] = (1.f - z) * n + z * sH[tid];
    }
    __syncthreads();
    if (tid < H_) {
      sH[tid] = sHn[tid];
      ys[((size_t)t * B_ + b) * H_ + tid] = sHn[tid];
    }
    __syncthreads();
  }
  if (dir == 0 && tid < H_) hT_f[b * H_ + tid] = sH[tid];
}

// ---------------------------------------------------------------------------
// K4: lin = concat(ys_f, ys_b) @ lin1_w^T + b  (small fp32 GEMM)
// ---------------------------------------------------------------------------
__global__ __launch_bounds__(256) void lin_kernel(
    const float* __restrict__ ys_f, const float* __restrict__ ys_b,
    const float* __restrict__ w, const float* __restrict__ bias,
    float* __restrict__ lin)
{
  const int idx = blockIdx.x * 256 + threadIdx.x;
  if (idx >= S_ * B_ * H_) return;
  const int row = idx / H_, j = idx % H_;
  const float* wj = w + (size_t)j * 2 * H_;
  const float* yf = ys_f + (size_t)row * H_;
  const float* yb = ys_b + (size_t)row * H_;
  float acc = bias[j];
  for (int k = 0; k < H_; ++k) acc += wj[k] * yf[k] + wj[H_ + k] * yb[k];
  lin[idx] = acc;
}

// ---------------------------------------------------------------------------
// K5: greedy decoder, one block per batch element, 32 sequential steps with
// argmax feedback. Writes [S, B, NC] log-probs.
// ---------------------------------------------------------------------------
__global__ __launch_bounds__(320) void dec_kernel(
    const float* __restrict__ lin, const float* __restrict__ hT_f,
    const float* __restrict__ emb, const float* __restrict__ wih,
    const float* __restrict__ whh, const float* __restrict__ bih,
    const float* __restrict__ bhh, const float* __restrict__ h2l_w,
    const float* __restrict__ h2l_b, float* __restrict__ out)
{
  __shared__ float sX[EMB_ + H_];
  __shared__ float sH[H_], sHn[H_];
  __shared__ float sGi[G3_], sGh[G3_];
  __shared__ float sLog[NC_];
  __shared__ float sLse;
  __shared__ int   sPrev;
  const int b = blockIdx.x, tid = threadIdx.x;
  if (tid < H_) sH[tid] = hT_f[b * H_ + tid];
  if (tid == 0) sPrev = 0;
  __syncthreads();

  for (int t = 0; t < S_; ++t) {
    const int prev = sPrev;
    if (tid < EMB_)            sX[tid] = emb[prev * EMB_ + tid];
    else if (tid < EMB_ + H_)  sX[tid] = lin[((size_t)t * B_ + b) * H_ + (tid - EMB_)];
    __syncthreads();

    if (tid < G3_) {
      const float* wi = wih + (size_t)tid * (EMB_ + H_);
      float a = bih[tid];
      for (int k = 0; k < EMB_ + H_; ++k) a += wi[k] * sX[k];
      sGi[tid] = a;
      const float* wh = whh + (size_t)tid * H_;
      float g = bhh[tid];
      for (int k = 0; k < H_; ++k) g += wh[k] * sH[k];
      sGh[tid] = g;
    }
    __syncthreads();

    if (tid < H_) {
      const float r = 1.f / (1.f + expf(-(sGi[tid]        + sGh[tid])));
      const float z = 1.f / (1.f + expf(-(sGi[H_ + tid]   + sGh[H_ + tid])));
      const float n = tanhf(sGi[2 * H_ + tid] + r * sGh[2 * H_ + tid]);
      sHn[tid] = (1.f - z) * n + z * sH[tid];
    }
    __syncthreads();
    if (tid < H_) sH[tid] = sHn[tid];
    __syncthreads();

    if (tid < NC_) {
      const float* wj = h2l_w + tid * H_;
      float a = h2l_b[tid];
      for (int k = 0; k < H_; ++k) a += wj[k] * sH[k];
      sLog[tid] = a;
    }
    __syncthreads();

    if (tid == 0) {
      float m = sLog[0]; int am = 0;
      for (int j = 1; j < NC_; ++j) if (sLog[j] > m) { m = sLog[j]; am = j; }
      float se = 0.f;
      for (int j = 0; j < NC_; ++j) se += expf(sLog[j] - m);
      sLse  = m + logf(se);
      sPrev = am;                      // first-max argmax, matches jnp.argmax
    }
    __syncthreads();
    if (tid < NC_) out[((size_t)t * B_ + b) * NC_ + tid] = sLog[tid] - sLse;
    __syncthreads();
  }
}

// ---------------------------------------------------------------------------
extern "C" void kernel_launch(void* const* d_in, const int* in_sizes, int n_in,
                              void* d_out, int out_size, void* d_ws, size_t ws_size,
                              hipStream_t stream)
{
  const float* hidden    = (const float*)d_in[0];
  const int*   cb        = (const int*)  d_in[1];
  const float* fc5_w     = (const float*)d_in[2];
  const float* fc5_b     = (const float*)d_in[3];
  const float* enc_wih_f = (const float*)d_in[4];
  const float* enc_whh_f = (const float*)d_in[5];
  const float* enc_bih_f = (const float*)d_in[6];
  const float* enc_bhh_f = (const float*)d_in[7];
  const float* enc_wih_b = (const float*)d_in[8];
  const float* enc_whh_b = (const float*)d_in[9];
  const float* enc_bih_b = (const float*)d_in[10];
  const float* enc_bhh_b = (const float*)d_in[11];
  const float* emb       = (const float*)d_in[12];
  const float* lin1_w    = (const float*)d_in[13];
  const float* lin1_b    = (const float*)d_in[14];
  const float* dec_wih   = (const float*)d_in[15];
  const float* dec_whh   = (const float*)d_in[16];
  const float* dec_bih   = (const float*)d_in[17];
  const float* dec_bhh   = (const float*)d_in[18];
  const float* h2l_w     = (const float*)d_in[19];
  const float* h2l_b     = (const float*)d_in[20];
  float* out = (float*)d_out;
  (void)in_sizes; (void)n_in; (void)out_size; (void)ws_size;

  char* ws = (char*)d_ws; size_t off = 0;
  auto carve = [&](size_t bytes) -> char* {
    char* p = ws + off; off += (bytes + 255) & ~(size_t)255; return p;
  };
  __bf16* doc_bf = (__bf16*)carve((size_t)S_ * B_ * D_ * 2);   // 3.0 MB
  __bf16* wf_bf  = (__bf16*)carve((size_t)G3_ * D_ * 2);       // 0.44 MB
  __bf16* wb_bf  = (__bf16*)carve((size_t)G3_ * D_ * 2);
  float*  gif    = (float*) carve((size_t)S_ * B_ * G3_ * 4);  // 2.34 MB
  float*  gib    = (float*) carve((size_t)S_ * B_ * G3_ * 4);
  float*  ysf    = (float*) carve((size_t)S_ * B_ * H_ * 4);
  float*  ysb    = (float*) carve((size_t)S_ * B_ * H_ * 4);
  float*  hTf    = (float*) carve((size_t)B_ * H_ * 4);
  float*  linb   = (float*) carve((size_t)S_ * B_ * H_ * 4);

  // K0: weights -> bf16
  {
    const int n = G3_ * D_;
    cvt_bf16_kernel<<<(n + 255) / 256, 256, 0, stream>>>(enc_wih_f, wf_bf, n);
    cvt_bf16_kernel<<<(n + 255) / 256, 256, 0, stream>>>(enc_wih_b, wb_bf, n);
  }
  // K1: clause pooling
  pool_kernel<<<B_ * S_, 256, 0, stream>>>(hidden, cb, fc5_w, fc5_b, doc_bf);
  // K2: WMMA input-transform GEMMs (both directions)
  {
    dim3 grid((S_ * B_ / 16) / 4, (G3_ + 15) / 16);   // 32 x 19
    gi_wmma_kernel<<<grid, 128, 0, stream>>>(doc_bf, wf_bf, enc_bih_f, gif);
    gi_wmma_kernel<<<grid, 128, 0, stream>>>(doc_bf, wb_bf, enc_bih_b, gib);
  }
  // K3: encoder recurrence, fwd+bwd concurrently
  {
    dim3 grid(B_, 2);
    const size_t smem = (size_t)30516 * 4;  // W_hh + h + gh + h_new
    enc_gru_kernel<<<grid, 320, smem, stream>>>(gif, gib, enc_whh_f, enc_whh_b,
                                                enc_bhh_f, enc_bhh_b, ysf, ysb, hTf);
  }
  // K4: lin projection
  {
    const int n = S_ * B_ * H_;
    lin_kernel<<<(n + 255) / 256, 256, 0, stream>>>(ysf, ysb, lin1_w, lin1_b, linb);
  }
  // K5: greedy decoder -> d_out [S, B, NC]
  dec_kernel<<<B_, 320, 0, stream>>>(linb, hTf, emb, dec_wih, dec_whh,
                                     dec_bih, dec_bhh, h2l_w, h2l_b, out);
}
